// NeuralSheaf_69217692942795
// MI455X (gfx1250) — compile-verified, matrix operand order checked by reading, not run on hardware
//
#include <hip/hip_runtime.h>
#include <hip/hip_bf16.h>

// ---------------------------------------------------------------------------
// NeuralSheaf on MI455X (gfx1250): f16 WMMA MLPs, TDM weight staging,
// async global->LDS edge gather, L2-resident fp32 atomic scatter.
//   B=2, N=50000, IN=64, HID=256, OUT=128, E=500000
// ---------------------------------------------------------------------------

#define GB 2
#define GN 50000
#define GIN 64
#define GHID 256
#define GOUT 128
#define GE 500000

typedef __attribute__((ext_vector_type(16))) _Float16     v16h;
typedef __attribute__((ext_vector_type(8)))  float        v8f;
typedef __attribute__((ext_vector_type(4)))  unsigned int v4u;
typedef __attribute__((ext_vector_type(8)))  int          v8i;
typedef __attribute__((ext_vector_type(4)))  int          v4i;

union V16 { v16h v; uint4 q[2]; };
union V8  { v8f  v; float f[8]; };

// ---------------------------------------------------------------------------
// TDM: 1-D copy (bytes multiple of 8, <= 64K units) global -> LDS via a 2-D
// tensor descriptor (D#), issued by one wave. CDNA5 ISA.
// ---------------------------------------------------------------------------
__device__ __forceinline__ void tdm_copy_to_lds(const void* gsrc, unsigned int ldsOff,
                                                unsigned int bytes) {
  unsigned long long ga = (unsigned long long)gsrc;
  unsigned int n8 = bytes >> 3;                       // units of data_size = 8B
  v4u g0;
  g0.x = 1u;                                          // count=1, user descriptor
  g0.y = ldsOff;                                      // lds_addr (bytes)
  g0.z = (unsigned int)ga;                            // global_addr[31:0]
  g0.w = (unsigned int)((ga >> 32) & 0x01FFFFFFu) | (2u << 30);  // addr[56:32] | type=2
  v8i g1;
  g1[0] = (int)(3u << 16);                            // wg_mask=0, data_size=3 (8B)
  g1[1] = (int)((n8 & 0xFFFFu) << 16);                // tensor_dim0[15:0]  (bits 79:48)
  g1[2] = (int)((n8 >> 16) | (1u << 16));             // tensor_dim0[31:16] | tensor_dim1=1
  g1[3] = (int)((n8 & 0xFFFFu) << 16);                // tile_dim0 (bits 127:112)
  g1[4] = 1;                                          // tile_dim1=1, tile_dim2=0
  g1[5] = (int)n8;                                    // tensor_dim0_stride[31:0]
  g1[6] = 0;                                          // stride0 hi / stride1 lo
  g1[7] = 0;
  v4i g2 = {0, 0, 0, 0};
  v4i g3 = {0, 0, 0, 0};
  asm volatile("tensor_load_to_lds %0, %1, %2, %3"
               :: "s"(g0), "s"(g1), "s"(g2), "s"(g3) : "memory");
  __builtin_amdgcn_s_wait_tensorcnt(0);
}

// A-fragment (16x32 f16, M=lane&15) from a row-major LDS tile.
// lanes 0-15 hold K = {0..7, 16..23}, lanes 16-31 hold K = {8..15, 24..31}.
__device__ __forceinline__ v16h ldsA(const _Float16* tile, int stride, int lane, int ks) {
  const int m    = lane & 15;
  const int base = (lane & 16) ? 8 : 0;
  const _Float16* r = tile + m * stride + ks * 32 + base;
  V16 u;
  u.q[0] = *(const uint4*)(r);
  u.q[1] = *(const uint4*)(r + 16);
  return u.v;
}

// B-fragment from pre-packed LDS weights: frag = ks*NT + ct, 16 contiguous f16/lane.
__device__ __forceinline__ v16h ldsB(const _Float16* wp, int frag, int lane) {
  const _Float16* r = wp + (((frag << 5) + lane) << 4);
  V16 u;
  u.q[0] = *(const uint4*)(r);
  u.q[1] = *(const uint4*)(r + 8);
  return u.v;
}

template<int KS>
__device__ __forceinline__ void loadA(v16h (&a)[KS], const _Float16* tile, int stride, int lane) {
#pragma unroll
  for (int k = 0; k < KS; ++k) a[k] = ldsA(tile, stride, lane, k);
}

// Preload all KS B-fragments, then issue WMMAs back-to-back (loads overlap WMMA).
template<int KS>
__device__ __forceinline__ v8f gemm_ct(const v16h (&a)[KS], const _Float16* wLds,
                                       int nt, int ct, int lane) {
  v16h b[KS];
#pragma unroll
  for (int k = 0; k < KS; ++k) b[k] = ldsB(wLds, k * nt + ct, lane);
  v8f acc = {0.f, 0.f, 0.f, 0.f, 0.f, 0.f, 0.f, 0.f};
#pragma unroll
  for (int k = 0; k < KS; ++k)
    acc = __builtin_amdgcn_wmma_f32_16x16x32_f16(
        false, a[k], false, b[k], (short)0, acc, false, false);
  return acc;
}

// ---------------------------------------------------------------------------
// Pack a row-major fp32 [K][Nc] weight into per-lane WMMA B-fragment order.
// ---------------------------------------------------------------------------
__global__ void pack_w(const float* __restrict__ W, _Float16* __restrict__ dst,
                       int Nc, int KS, int NT) {
  int i = blockIdx.x * blockDim.x + threadIdx.x;
  int total = KS * NT * 32 * 16;
  if (i >= total) return;
  int j    = i & 15;
  int lane = (i >> 4) & 31;
  int rest = i >> 9;
  int ct   = rest % NT;
  int ks   = rest / NT;
  int base = (lane < 16) ? 0 : 8;
  int k    = ks * 32 + ((j < 8) ? (base + j) : (16 + base + (j - 8)));
  int n    = ct * 16 + (lane & 15);
  dst[i] = (_Float16)W[(size_t)k * Nc + n];
}

// x fp32 -> f16 copy (for async LDS gather in the edge kernel)
__global__ void convert_x(const float* __restrict__ x, _Float16* __restrict__ xh, int n8) {
  int i = blockIdx.x * blockDim.x + threadIdx.x;
  if (i >= n8) return;
  const float4* s = (const float4*)(x + (size_t)i * 8);
  float4 f0 = s[0], f1 = s[1];
  union { _Float16 h[8]; uint4 q; } o;
  o.h[0] = (_Float16)f0.x; o.h[1] = (_Float16)f0.y;
  o.h[2] = (_Float16)f0.z; o.h[3] = (_Float16)f0.w;
  o.h[4] = (_Float16)f1.x; o.h[5] = (_Float16)f1.y;
  o.h[6] = (_Float16)f1.z; o.h[7] = (_Float16)f1.w;
  *(uint4*)(xh + (size_t)i * 8) = o.q;
}

// ---------------------------------------------------------------------------
// Node MLP: h = relu(x @ nw1 + nb1) @ nw2 + nb2       rows = B*N = 100000
// ---------------------------------------------------------------------------
__global__ __launch_bounds__(256)
void node_kernel(const float* __restrict__ x,
                 const _Float16* __restrict__ w1p, const float* __restrict__ b1,
                 const _Float16* __restrict__ w2p, const float* __restrict__ b2,
                 float* __restrict__ h) {
  extern __shared__ char smem_raw[];
  _Float16* sW1  = (_Float16*)smem_raw;          // 2*16*512 = 16384 elems (32 KB)
  _Float16* sW2  = sW1 + 16384;                  // 8*8*512  = 32768 elems (64 KB)
  _Float16* sInp = sW2 + 32768;                  // 8 waves * 16*64
  _Float16* sHid = sInp + 8 * 1024;              // 8 waves * 16*256

  if (threadIdx.x < 32) {                        // wave 0: TDM-stage both weight panels
    tdm_copy_to_lds(w1p, (unsigned int)(unsigned long long)sW1, (16384 + 32768) * 2);
  }
  __syncthreads();

  const int wave = threadIdx.x >> 5;
  const int lane = threadIdx.x & 31;
  _Float16* inp = sInp + wave * 1024;
  _Float16* hid = sHid + wave * 4096;
  const int n15 = lane & 15;
  const int mb  = (lane & 16) ? 8 : 0;

  const int nTiles = (GB * GN) / 16;             // 6250
  for (int tile = blockIdx.x * 8 + wave; tile < nTiles; tile += gridDim.x * 8) {
    { // stage 16 rows of x (fp32 -> f16)
      int m = lane >> 1, half = lane & 1;
      const float4* xr = (const float4*)(x + (size_t)(tile * 16 + m) * GIN + half * 32);
      _Float16* drow = inp + m * GIN + half * 32;
#pragma unroll
      for (int t = 0; t < 8; ++t) {
        float4 f = xr[t];
        drow[4 * t + 0] = (_Float16)f.x; drow[4 * t + 1] = (_Float16)f.y;
        drow[4 * t + 2] = (_Float16)f.z; drow[4 * t + 3] = (_Float16)f.w;
      }
    }
    // layer 1: 16x64 @ 64x256, KS=2, NT=16
    v16h a[2]; loadA<2>(a, inp, GIN, lane);
#pragma unroll
    for (int ct = 0; ct < 16; ++ct) {
      V8 acc; acc.v = gemm_ct<2>(a, sW1, 16, ct, lane);
      int n = ct * 16 + n15;
      float bias = b1[n];
#pragma unroll
      for (int r = 0; r < 8; ++r) {
        float v = acc.f[r] + bias;
        hid[(mb + r) * GHID + n] = (_Float16)(v > 0.f ? v : 0.f);
      }
    }
    // layer 2: 16x256 @ 256x128, KS=8, NT=8 -> store h
    v16h c[8]; loadA<8>(c, hid, GHID, lane);
#pragma unroll
    for (int ct = 0; ct < 8; ++ct) {
      V8 acc; acc.v = gemm_ct<8>(c, sW2, 8, ct, lane);
      int n = ct * 16 + n15;
      float bias = b2[n];
#pragma unroll
      for (int r = 0; r < 8; ++r)
        h[(size_t)(tile * 16 + mb + r) * GOUT + n] = acc.f[r] + bias;
    }
  }
}

// ---------------------------------------------------------------------------
// Edge MLP + scatter. Gather uses async global->LDS b128 (ASYNCcnt) from xh.
// hbase[] doubles as: byte offset into xh (row=128B) and f32-elem offset into h.
// ---------------------------------------------------------------------------
__global__ __launch_bounds__(256)
void edge_kernel(const _Float16* __restrict__ xh, const int* __restrict__ edges,
                 const _Float16* __restrict__ w1p, const float* __restrict__ b1,
                 const _Float16* __restrict__ w2p, const float* __restrict__ b2,
                 float* __restrict__ h) {
  extern __shared__ char smem_raw[];
  _Float16* sW1  = (_Float16*)smem_raw;          // 4*16*512 = 32768 elems (64 KB)
  _Float16* sW2  = sW1 + 32768;                  // 8*8*512  = 32768 elems (64 KB)
  _Float16* sInp = sW2 + 32768;                  // 8 waves * 16*128
  _Float16* sHid = sInp + 8 * 2048;              // 8 waves * 16*256
  int*      sIdx = (int*)(sHid + 8 * 4096);      // 8 waves * 32 row bases

  if (threadIdx.x < 32) {                        // wave 0: TDM-stage both weight panels
    tdm_copy_to_lds(w1p, (unsigned int)(unsigned long long)sW1, (32768 + 32768) * 2);
  }
  __syncthreads();

  const int wave = threadIdx.x >> 5;
  const int lane = threadIdx.x & 31;
  _Float16* inp = sInp + wave * 2048;
  _Float16* hid = sHid + wave * 4096;
  int* hbase = sIdx + wave * 32;
  const unsigned int inpOff = (unsigned int)(unsigned long long)inp;  // LDS byte addr
  const int n15 = lane & 15;
  const int mb  = (lane & 16) ? 8 : 0;

  const int nTiles = (GB * GE) / 16;             // 62500
  for (int tile = blockIdx.x * 8 + wave; tile < nTiles; tile += gridDim.x * 8) {
    { // edge-index stage: lane -> row m = lane>>1, endpoint half = lane&1
      int m = lane >> 1, half = lane & 1;
      int r = tile * 16 + m;
      int b = (r >= GE) ? 1 : 0;
      int e = r - b * GE;
      int src = edges[2 * e + half];              // half0 = u, half1 = v
      hbase[2 * m + half] = (b * GN + src) * GOUT; // = xh byte offset = h elem offset
    }
    { // async gather: 16 rows x 256 B from xh -> inp (8 x b128 per lane)
      asm volatile("s_wait_dscnt 0x0" ::: "memory");  // inp reads of prev tile done
#pragma unroll
      for (int i = 0; i < 8; ++i) {
        int g    = i * 32 + lane;                 // 16B-chunk id, 0..255
        int m    = g >> 4;
        int ch   = g & 15;
        int half = ch >> 3;
        int sub  = ch & 7;
        unsigned long long ga = (unsigned long long)xh
                              + (unsigned int)hbase[2 * m + half]
                              + (unsigned int)(sub * 16);
        unsigned int la = inpOff + (unsigned int)(m * 256 + half * 128 + sub * 16);
        asm volatile("global_load_async_to_lds_b128 %0, %1, off"
                     :: "v"(la), "v"(ga) : "memory");
      }
      asm volatile("s_wait_asynccnt 0x0" ::: "memory");
    }
    // layer 1: 16x128 @ 128x256, KS=4, NT=16
    v16h a[4]; loadA<4>(a, inp, 128, lane);
#pragma unroll
    for (int ct = 0; ct < 16; ++ct) {
      V8 acc; acc.v = gemm_ct<4>(a, sW1, 16, ct, lane);
      int n = ct * 16 + n15;
      float bias = b1[n];
#pragma unroll
      for (int r = 0; r < 8; ++r) {
        float v = acc.f[r] + bias;
        hid[(mb + r) * GHID + n] = (_Float16)(v > 0.f ? v : 0.f);
      }
    }
    // layer 2: 16x256 @ 256x128, KS=8, NT=8 -> scatter-add to both endpoints
    v16h c[8]; loadA<8>(c, hid, GHID, lane);
#pragma unroll
    for (int ct = 0; ct < 8; ++ct) {
      V8 acc; acc.v = gemm_ct<8>(c, sW2, 8, ct, lane);
      int n = ct * 16 + n15;
      float bias = b2[n];
#pragma unroll
      for (int r = 0; r < 8; ++r) {
        float v = acc.f[r] + bias;
        int m = mb + r;
        unsafeAtomicAdd(h + hbase[2 * m + 0] + n, v);   // global_atomic_add_f32
        unsafeAtomicAdd(h + hbase[2 * m + 1] + n, v);
      }
    }
  }
}

// ---------------------------------------------------------------------------
// Recon: out = h @ dw + db     (100000 x 128) @ (128 x 64)
// ---------------------------------------------------------------------------
__global__ __launch_bounds__(256)
void recon_kernel(const float* __restrict__ h,
                  const _Float16* __restrict__ dwp, const float* __restrict__ db,
                  float* __restrict__ out) {
  extern __shared__ char smem_raw[];
  _Float16* sW   = (_Float16*)smem_raw;          // 4*4*512 = 8192 elems (16 KB)
  _Float16* sInp = sW + 8192;                    // 8 waves * 16*128

  if (threadIdx.x < 32) {
    tdm_copy_to_lds(dwp, (unsigned int)(unsigned long long)sW, 8192 * 2);
  }
  __syncthreads();

  const int wave = threadIdx.x >> 5;
  const int lane = threadIdx.x & 31;
  _Float16* inp = sInp + wave * 2048;
  const int n15 = lane & 15;
  const int mb  = (lane & 16) ? 8 : 0;

  const int nTiles = (GB * GN) / 16;             // 6250
  for (int tile = blockIdx.x * 8 + wave; tile < nTiles; tile += gridDim.x * 8) {
    { // stage h rows fp32 -> f16
      int m = lane >> 1, half = lane & 1;
      const float4* hr = (const float4*)(h + (size_t)(tile * 16 + m) * GOUT + half * 64);
      _Float16* drow = inp + m * GOUT + half * 64;
#pragma unroll
      for (int t = 0; t < 16; ++t) {
        float4 f = hr[t];
        drow[4 * t + 0] = (_Float16)f.x; drow[4 * t + 1] = (_Float16)f.y;
        drow[4 * t + 2] = (_Float16)f.z; drow[4 * t + 3] = (_Float16)f.w;
      }
    }
    v16h a[4]; loadA<4>(a, inp, GOUT, lane);
#pragma unroll
    for (int ct = 0; ct < 4; ++ct) {
      V8 acc; acc.v = gemm_ct<4>(a, sW, 4, ct, lane);
      int n = ct * 16 + n15;
      float bias = db[n];
#pragma unroll
      for (int r = 0; r < 8; ++r)
        out[(size_t)(tile * 16 + mb + r) * GIN + n] = acc.f[r] + bias;
    }
  }
}

// ---------------------------------------------------------------------------
extern "C" void kernel_launch(void* const* d_in, const int* in_sizes, int n_in,
                              void* d_out, int out_size, void* d_ws, size_t ws_size,
                              hipStream_t stream) {
  const float* x   = (const float*)d_in[0];
  const int*  edges= (const int*)  d_in[1];
  const float* ew1 = (const float*)d_in[2];
  const float* eb1 = (const float*)d_in[3];
  const float* ew2 = (const float*)d_in[4];
  const float* eb2 = (const float*)d_in[5];
  const float* nw1 = (const float*)d_in[6];
  const float* nb1 = (const float*)d_in[7];
  const float* nw2 = (const float*)d_in[8];
  const float* nb2 = (const float*)d_in[9];
  const float* dw  = (const float*)d_in[10];
  const float* db  = (const float*)d_in[11];
  float* out = (float*)d_out;

  char* ws = (char*)d_ws;
  float* h = (float*)ws;                              // 51.2 MB
  size_t off = (size_t)GB * GN * GOUT * sizeof(float);
  _Float16* ew1p = (_Float16*)(ws + off); off += 4 * 16 * 512 * 2;  // 64 KB (contiguous
  _Float16* ew2p = (_Float16*)(ws + off); off += 8 * 8  * 512 * 2;  //  with ew1p for TDM)
  _Float16* nw1p = (_Float16*)(ws + off); off += 2 * 16 * 512 * 2;  // 32 KB (contiguous
  _Float16* nw2p = (_Float16*)(ws + off); off += 8 * 8  * 512 * 2;  //  with nw1p for TDM)
  _Float16* dwp  = (_Float16*)(ws + off); off += 4 * 4  * 512 * 2;  // 16 KB
  _Float16* xh   = (_Float16*)(ws + off); off += (size_t)GB * GN * GIN * 2; // 12.8 MB

  // weight packing + x conversion
  pack_w<<<(4 * 16 * 512 + 255) / 256, 256, 0, stream>>>(ew1, ew1p, GHID, 4, 16);
  pack_w<<<(8 * 8  * 512 + 255) / 256, 256, 0, stream>>>(ew2, ew2p, GOUT, 8, 8);
  pack_w<<<(2 * 16 * 512 + 255) / 256, 256, 0, stream>>>(nw1, nw1p, GHID, 2, 16);
  pack_w<<<(8 * 8  * 512 + 255) / 256, 256, 0, stream>>>(nw2, nw2p, GOUT, 8, 8);
  pack_w<<<(4 * 4  * 512 + 255) / 256, 256, 0, stream>>>(dw,  dwp,  GIN,  4, 4);
  const int n8 = GB * GN * GIN / 8;                   // 800000
  convert_x<<<(n8 + 255) / 256, 256, 0, stream>>>(x, xh, n8);

  // node MLP -> h
  size_t nodeLds = (size_t)(16384 + 32768 + 8 * 1024 + 8 * 4096) * 2;            // 180224 B
  node_kernel<<<512, 256, nodeLds, stream>>>(x, nw1p, nb1, nw2p, nb2, h);

  // edge MLP + scatter-add into h
  size_t edgeLds = (size_t)(32768 + 32768 + 8 * 2048 + 8 * 4096) * 2 + 8 * 32 * 4; // 230400 B
  edge_kernel<<<1024, 256, edgeLds, stream>>>(xh, edges, ew1p, eb1, ew2p, eb2, h);

  // recon = h @ dw + db -> d_out
  size_t reconLds = (size_t)(8192 + 8 * 2048) * 2;                               // 49152 B
  recon_kernel<<<512, 256, reconLds, stream>>>(h, dwp, db, out);
}
